// CrossAttentionLayer_72404558676374
// MI455X (gfx1250) — matrix operand order repaired and checked by reading, not compiled, not run on hardware
//
#include <hip/hip_runtime.h>
#include <hip/hip_bf16.h>

// ---- raw clang ext-vector types (avoid HIP vector classes in unions) ----
typedef __attribute__((ext_vector_type(16))) __bf16        v16bf;
typedef __attribute__((ext_vector_type(8)))  float         v8f;
typedef __attribute__((ext_vector_type(4))) unsigned int   u32x4;
typedef __attribute__((ext_vector_type(4)))  float         f32x4;

union FragBF { v16bf v; u32x4 q[2]; unsigned short u[16]; };
union Pack8  { u32x4 q;  unsigned short u[8]; };

#ifndef __has_builtin
#define __has_builtin(x) 0
#endif

// CDNA5 async global->LDS copy (ASYNCcnt path), via inline asm per
// CDNA5 ISA / 15.18.3 (GVS mode: VDST = per-lane LDS byte
// address, VADDR = u32 byte offset, SADDR = 64-bit base in SGPR pair).
// Set to 0 to fall back to a plain VGPR round-trip copy.
#define USE_ASYNC_LDS 1

__device__ __forceinline__ void async_wait0() {
#if __has_builtin(__builtin_amdgcn_s_wait_asynccnt)
  __builtin_amdgcn_s_wait_asynccnt(0);
#else
  asm volatile("s_wait_asynccnt 0x0" ::: "memory");
#endif
}

// 16-byte async copy: global (base + byteOff) -> LDS at generic pointer `lds`.
__device__ __forceinline__ void cp16_g2lds(const void* gbase, unsigned byteOff,
                                           void* lds) {
#if USE_ASYNC_LDS
  // LDS aperture keeps the LDS byte offset in addr[31:0] (ISA 10.2).
  const unsigned laddr = (unsigned)(unsigned long long)lds;
  asm volatile("global_load_async_to_lds_b128 %0, %1, %2"
               :: "v"(laddr), "v"(byteOff),
                  "s"((unsigned long long)gbase)
               : "memory");
#else
  *(u32x4*)lds = *(const u32x4*)((const char*)gbase + byteOff);
#endif
}

__device__ __forceinline__ unsigned short f2bf(float f) {
  unsigned int u = __float_as_uint(f);
  unsigned int r = u + 0x7FFFu + ((u >> 16) & 1u);   // RNE
  return (unsigned short)(r >> 16);
}

// =====================================================================
// Generic GEMM:  C[S,F] = A[S,K] @ W[K,F]   (A fp32 or bf16; W fp32)
// bf16 WMMA, f32 accumulate. Tile 64x128, BK=32, 256 threads (8 waves),
// each wave owns a 32x32 quad of C (4 WMMAs / K-step).
// =====================================================================
template<bool A_BF16, bool OUT_BF16>
__global__ __launch_bounds__(256)
void gemm_wmma_kernel(const void* __restrict__ Ap, const float* __restrict__ W,
                      void* __restrict__ Cp, const float* __restrict__ bias,
                      int S, int K, int F)
{
  __shared__ unsigned short As[64][40];    // [m][k]   (80B rows -> 16B aligned)
  __shared__ unsigned short Bs[128][40];   // transposed: [n][k]

  const int t    = threadIdx.x;
  const int lane = t & 31;
  const int wave = t >> 5;
  const int l16  = lane & 15;
  const int hl   = lane >> 4;       // lane half (0/1)
  const int wn   = wave & 3;        // 32-col slab 0..3
  const int wm   = wave >> 2;       // 32-row slab 0..1

  const int fBlk = blockIdx.x * 128;
  const int sBlk = blockIdx.y * 64;

  v8f c00 = {0,0,0,0,0,0,0,0}, c01 = {0,0,0,0,0,0,0,0};
  v8f c10 = {0,0,0,0,0,0,0,0}, c11 = {0,0,0,0,0,0,0,0};

  const int aRow = t >> 2, aCol = (t & 3) * 8;    // 64 rows x 32 cols,  8/thread
  const int wRow = t >> 3, wCol = (t & 7) * 16;   // 32 rows x 128 cols, 16/thread

  for (int k0 = 0; k0 < K; k0 += 32) {
    // ---- stage A tile ----
    if (A_BF16) {
      const size_t elt = (size_t)(sBlk + aRow) * K + k0 + aCol;
      cp16_g2lds(Ap, (unsigned)(elt * 2), &As[aRow][aCol]);   // async copy
    } else {
      const float* A = (const float*)Ap;
      const float* p = &A[(size_t)(sBlk + aRow) * K + k0 + aCol];
      f32x4 f0 = *(const f32x4*)p;
      f32x4 f1 = *(const f32x4*)(p + 4);
      Pack8 pa;
      #pragma unroll
      for (int e = 0; e < 4; ++e) { pa.u[e] = f2bf(f0[e]); pa.u[4 + e] = f2bf(f1[e]); }
      *(u32x4*)&As[aRow][aCol] = pa.q;
    }

    // ---- stage W tile transposed (f32 -> bf16) ----
    {
      const float* p = &W[(size_t)(k0 + wRow) * F + fBlk + wCol];
      #pragma unroll
      for (int e2 = 0; e2 < 4; ++e2) {
        f32x4 f = *(const f32x4*)(p + e2 * 4);
        #pragma unroll
        for (int e = 0; e < 4; ++e) Bs[wCol + e2 * 4 + e][wRow] = f2bf(f[e]);
      }
    }
#if USE_ASYNC_LDS
    if (A_BF16) async_wait0();
#endif
    __syncthreads();

    // ---- fragments (documented 16-bit WMMA layouts) ----
    FragBF a0, a1, b0, b1;
    a0.q[0] = *(const u32x4*)&As[wm * 32 + l16][hl * 8];
    a0.q[1] = *(const u32x4*)&As[wm * 32 + l16][16 + hl * 8];
    a1.q[0] = *(const u32x4*)&As[wm * 32 + 16 + l16][hl * 8];
    a1.q[1] = *(const u32x4*)&As[wm * 32 + 16 + l16][16 + hl * 8];
    b0.q[0] = *(const u32x4*)&Bs[wn * 32 + l16][hl * 8];
    b0.q[1] = *(const u32x4*)&Bs[wn * 32 + l16][16 + hl * 8];
    b1.q[0] = *(const u32x4*)&Bs[wn * 32 + 16 + l16][hl * 8];
    b1.q[1] = *(const u32x4*)&Bs[wn * 32 + 16 + l16][16 + hl * 8];

    c00 = __builtin_amdgcn_wmma_f32_16x16x32_bf16(false, a0.v, false, b0.v, (short)0, c00, false, false);
    c01 = __builtin_amdgcn_wmma_f32_16x16x32_bf16(false, a0.v, false, b1.v, (short)0, c01, false, false);
    c10 = __builtin_amdgcn_wmma_f32_16x16x32_bf16(false, a1.v, false, b0.v, (short)0, c10, false, false);
    c11 = __builtin_amdgcn_wmma_f32_16x16x32_bf16(false, a1.v, false, b1.v, (short)0, c11, false, false);
    __syncthreads();
  }

  // ---- epilogue ----
  const int n0 = fBlk + wn * 32 + l16;
  const float bv0 = (!OUT_BF16 && bias) ? bias[n0] : 0.0f;
  const float bv1 = (!OUT_BF16 && bias) ? bias[n0 + 16] : 0.0f;
  #pragma unroll
  for (int r = 0; r < 8; ++r) {
    const int m0 = sBlk + wm * 32 + 8 * hl + r;
    const int m1 = m0 + 16;
    if (OUT_BF16) {
      unsigned short* C = (unsigned short*)Cp;
      C[(size_t)m0 * F + n0]      = f2bf(c00[r]);
      C[(size_t)m0 * F + n0 + 16] = f2bf(c01[r]);
      C[(size_t)m1 * F + n0]      = f2bf(c10[r]);
      C[(size_t)m1 * F + n0 + 16] = f2bf(c11[r]);
    } else {
      float* C = (float*)Cp;
      C[(size_t)m0 * F + n0]      = c00[r] + bv0;
      C[(size_t)m0 * F + n0 + 16] = c01[r] + bv1;
      C[(size_t)m1 * F + n0]      = c10[r] + bv0;
      C[(size_t)m1 * F + n0 + 16] = c11[r] + bv1;
    }
  }
}

// =====================================================================
// Fused attention: per (b, head, 64-query block); 4 waves, 16 queries each.
// sim = qK^T / 8, mask->-5, clip[-5,5], exp (clip bounds make running-max
// unnecessary), P@V accumulated in f32, row-sum normalization at the end.
// K chunk staged via async global->LDS; V staged transposed.
// =====================================================================
__global__ __launch_bounds__(128)
void attn_wmma_kernel(const unsigned short* __restrict__ qbf,   // [B*1024][1024]
                      const unsigned short* __restrict__ kvbf,  // [B*4096][2048] (k|v)
                      const unsigned char*  __restrict__ mask,  // [B*4096]
                      unsigned short*       __restrict__ obf)   // [B*1024][1024]
{
  __shared__ unsigned short Ks[32][72];      // [j][d]
  __shared__ unsigned short Vs[64][40];      // transposed: [d][j]
  __shared__ unsigned short Ps[4][16][40];   // per-wave P transpose buffer

  const int t    = threadIdx.x;
  const int lane = t & 31;
  const int wave = t >> 5;
  const int l16  = lane & 15;
  const int hl   = lane >> 4;

  const int blk   = blockIdx.x;        // 1024 = B(4) * H(16) * (N/64=16)
  const int ntile = blk & 15;
  const int head  = (blk >> 4) & 15;
  const int b     = blk >> 8;

  const int n0 = ntile * 64 + wave * 16;

  // q A-fragments for the two K-steps (d 0..31, 32..63), straight from global
  FragBF aq0, aq1;
  {
    const size_t qrow = (size_t)(b * 1024 + n0 + l16) * 1024 + head * 64;
    aq0.q[0] = *(const u32x4*)&qbf[qrow + hl * 8];
    aq0.q[1] = *(const u32x4*)&qbf[qrow + 16 + hl * 8];
    aq1.q[0] = *(const u32x4*)&qbf[qrow + 32 + hl * 8];
    aq1.q[1] = *(const u32x4*)&qbf[qrow + 48 + hl * 8];
  }

  v8f o[4];
  #pragma unroll
  for (int dt = 0; dt < 4; ++dt) o[dt] = (v8f){0,0,0,0,0,0,0,0};
  v8f rs = {0,0,0,0,0,0,0,0};

  const size_t kvbase = (size_t)b * 4096 * 2048 + (size_t)head * 64;
  const int    mbase  = b * 4096;

  for (int j0 = 0; j0 < 4096; j0 += 32) {
    // ---- cooperative K/V chunk staging (32 rows x 64 cols, bf16) ----
    #pragma unroll
    for (int i = 0; i < 2; ++i) {
      const int idx = i * 128 + t;
      const int row = idx >> 3;
      const int cq  = (idx & 7) * 8;
      const size_t base = kvbase + (size_t)(j0 + row) * 2048 + cq;
      cp16_g2lds(kvbf, (unsigned)(base * 2), &Ks[row][cq]);  // K row-major (async)
      Pack8 pv; pv.q = *(const u32x4*)&kvbf[base + 1024];    // V transposed
      #pragma unroll
      for (int e = 0; e < 8; ++e) Vs[cq + e][row] = pv.u[e];
    }
    if (j0 + 32 < 4096)   // hint next chunk toward L2
      __builtin_prefetch(&kvbf[kvbase + (size_t)(j0 + 32 + (t >> 2)) * 2048], 0, 1);
#if USE_ASYNC_LDS
    async_wait0();
#endif
    __syncthreads();

    // ---- sim tiles (two 16-col j tiles), exp, stash P in LDS ----
    #pragma unroll
    for (int jt = 0; jt < 2; ++jt) {
      FragBF bk0, bk1;
      const int jr = jt * 16 + l16;
      bk0.q[0] = *(const u32x4*)&Ks[jr][hl * 8];
      bk0.q[1] = *(const u32x4*)&Ks[jr][16 + hl * 8];
      bk1.q[0] = *(const u32x4*)&Ks[jr][32 + hl * 8];
      bk1.q[1] = *(const u32x4*)&Ks[jr][48 + hl * 8];

      v8f c = {0,0,0,0,0,0,0,0};
      c = __builtin_amdgcn_wmma_f32_16x16x32_bf16(false, aq0.v, false, bk0.v, (short)0, c, false, false);
      c = __builtin_amdgcn_wmma_f32_16x16x32_bf16(false, aq1.v, false, bk1.v, (short)0, c, false, false);

      const int   j    = j0 + jt * 16 + l16;
      const float keep = mask[mbase + j] ? 1.0f : 0.0f;
      #pragma unroll
      for (int r = 0; r < 8; ++r) {
        float val = c[r] * 0.125f;                    // DH^-0.5
        val = fminf(fmaxf(val, -5.0f), 5.0f);
        val = keep * val + (1.0f - keep) * -5.0f;     // masked -> exactly -5
        const float p = __expf(val);
        rs[r] += p;
        Ps[wave][8 * hl + r][jt * 16 + l16] = f2bf(p);
      }
    }

    // ---- P A-fragment (same-wave LDS store->load is in order) ----
    FragBF aP;
    aP.q[0] = *(const u32x4*)&Ps[wave][l16][hl * 8];
    aP.q[1] = *(const u32x4*)&Ps[wave][l16][16 + hl * 8];

    #pragma unroll
    for (int dt = 0; dt < 4; ++dt) {
      FragBF bv;
      bv.q[0] = *(const u32x4*)&Vs[dt * 16 + l16][hl * 8];
      bv.q[1] = *(const u32x4*)&Vs[dt * 16 + l16][16 + hl * 8];
      o[dt] = __builtin_amdgcn_wmma_f32_16x16x32_bf16(false, aP.v, false, bv.v, (short)0, o[dt], false, false);
    }
    __syncthreads();   // protect Ks/Vs before next staging
  }

  // ---- row sums: reduce across the 16-lane C-layout groups ----
  float rsum[8];
  #pragma unroll
  for (int r = 0; r < 8; ++r) {
    float s = rs[r];
    s += __shfl_xor(s, 1);
    s += __shfl_xor(s, 2);
    s += __shfl_xor(s, 4);
    s += __shfl_xor(s, 8);
    rsum[r] = s;
  }

  // ---- normalize + store bf16 in [b, n, h*64+d] layout ----
  #pragma unroll
  for (int dt = 0; dt < 4; ++dt) {
    #pragma unroll
    for (int r = 0; r < 8; ++r) {
      const size_t row = (size_t)(b * 1024 + n0 + 8 * hl + r);
      obf[row * 1024 + head * 64 + dt * 16 + l16] = f2bf(o[dt][r] / rsum[r]);
    }
  }
}

// =====================================================================
// Host launch
// =====================================================================
extern "C" void kernel_launch(void* const* d_in, const int* in_sizes, int n_in,
                              void* d_out, int out_size, void* d_ws, size_t ws_size,
                              hipStream_t stream) {
  const float*         x    = (const float*)d_in[0];          // [4,1024,1024]
  const float*         ctx  = (const float*)d_in[1];          // [4,4096,1024]
  const unsigned char* mask = (const unsigned char*)d_in[2];  // [4,4096] bool
  const float*         Wq   = (const float*)d_in[3];          // [1024,1024]
  const float*         Wkv  = (const float*)d_in[4];          // [1024,2048]
  const float*         Wo   = (const float*)d_in[5];          // [1024,1024]
  const float*         bo   = (const float*)d_in[6];          // [1024]
  float*               out  = (float*)d_out;                  // [4,1024,1024]

  // workspace: q (8MB) | kv (64MB) | attn (8MB), all bf16
  unsigned short* qbf   = (unsigned short*)d_ws;
  unsigned short* kvbf  = qbf  + (size_t)4096 * 1024;
  unsigned short* attnb = kvbf + (size_t)16384 * 2048;

  // q = x @ Wq                    [4096,1024] x [1024,1024] -> bf16
  gemm_wmma_kernel<false, true><<<dim3(1024 / 128, 4096 / 64), 256, 0, stream>>>(
      x, Wq, qbf, nullptr, 4096, 1024, 1024);
  // kv = context @ Wkv            [16384,1024] x [1024,2048] -> bf16
  gemm_wmma_kernel<false, true><<<dim3(2048 / 128, 16384 / 64), 256, 0, stream>>>(
      ctx, Wkv, kvbf, nullptr, 16384, 1024, 2048);
  // fused masked softmax attention
  attn_wmma_kernel<<<dim3(1024), 128, 0, stream>>>(qbf, kvbf, mask, attnb);
  // out = attn @ Wo + bo          [4096,1024] x [1024,1024] -> f32
  gemm_wmma_kernel<true, false><<<dim3(1024 / 128, 4096 / 64), 256, 0, stream>>>(
      attnb, Wo, out, bo, 4096, 1024, 1024);
}
